// PairBiasedAttention_58213986730471
// MI455X (gfx1250) — compile-verified, matrix operand order checked by reading, not run on hardware
//
#include <hip/hip_runtime.h>

#define DM   512   // d_model
#define NSEQ 512   // sequence length
#define BATCH 2
#define NH   8     // heads
#define DH   64    // head dim
#define DP   128   // pair dim

typedef __bf16 bf16_t;
typedef __attribute__((ext_vector_type(16))) __bf16 v16bf;
typedef __attribute__((ext_vector_type(8)))  float  v8f;

// K index inside a 32-wide chunk for A/B fragments of v_wmma_*_16x16x32_bf16.
// Lane group g = lane/16, element e = 0..15 within the 8-VGPR fragment:
//   g=0: K = {0..7, 16..23};  g=1: K = {8..15, 24..31}
__device__ __forceinline__ int kmap(int e, int g) {
    return (e < 8 ? e : e + 8) + 8 * g;
}

// ---------------------------------------------------------------------------
// 1) Q/K/V projection: out = x @ W^T   (1024x512 @ 512x512^T), bf16 WMMA
// ---------------------------------------------------------------------------
__global__ void pba_qkv_gemm(const float* __restrict__ x,
                             const float* __restrict__ Wq,
                             const float* __restrict__ Wk,
                             const float* __restrict__ Wv,
                             float* __restrict__ qf,
                             float* __restrict__ kf,
                             float* __restrict__ vf)
{
    const int wid  = threadIdx.x >> 5;
    const int lane = threadIdx.x & 31;
    const int l16  = lane & 15;
    const int g    = lane >> 4;

    const float* W; float* out;
    switch (blockIdx.z) {
        case 0:  W = Wq; out = qf; break;
        case 1:  W = Wk; out = kf; break;
        default: W = Wv; out = vf; break;
    }

    const int TILES_J = DM / 16;         // 32
    int t  = blockIdx.x * (blockDim.x >> 5) + wid;
    int ti = t / TILES_J, tj = t % TILES_J;
    if (ti >= (BATCH * NSEQ) / 16) return;

    const int arow = ti * 16 + l16;
    const int bcol = tj * 16 + l16;

    v8f acc = {};
    for (int k0 = 0; k0 < DM; k0 += 32) {
        v16bf a, b;
#pragma unroll
        for (int e = 0; e < 16; ++e) {
            a[e] = (bf16_t)x[(size_t)arow * DM + k0 + kmap(e, g)];
            b[e] = (bf16_t)W[(size_t)bcol * DM + k0 + kmap(e, g)]; // B[k,j]=W[j,k]
        }
        acc = __builtin_amdgcn_wmma_f32_16x16x32_bf16(
                  false, a, false, b, (short)0, acc, false, false);
    }
#pragma unroll
    for (int r = 0; r < 8; ++r)
        out[(size_t)(ti * 16 + r + 8 * g) * DM + tj * 16 + l16] = acc[r];
}

// ---------------------------------------------------------------------------
// 2) Per-head LayerNorm (f32) + repack to bf16.
//    q,k -> [B,H,N,DH] row-major;  v -> TRANSPOSED [B,H,DH,N] so that the
//    P@V B-fragments in attention read contiguous 16B groups along m.
// ---------------------------------------------------------------------------
__global__ void pba_ln_pack(const float* __restrict__ qf,
                            const float* __restrict__ kf,
                            const float* __restrict__ vf,
                            const float* __restrict__ qg, const float* __restrict__ qb,
                            const float* __restrict__ kg, const float* __restrict__ kb,
                            bf16_t* __restrict__ qbf,
                            bf16_t* __restrict__ kbf,
                            bf16_t* __restrict__ vtbf)
{
    const int wid  = threadIdx.x >> 5;
    const int lane = threadIdx.x & 31;
    int idx = blockIdx.x * (blockDim.x >> 5) + wid;     // over B*N*H
    if (idx >= BATCH * NSEQ * NH) return;
    int h  = idx % NH;
    int bn = idx / NH;
    int n  = bn % NSEQ, b = bn / NSEQ;
    int bh = b * NH + h;

    const float* src; const float* gamma; const float* beta; bool do_ln;
    switch (blockIdx.y) {
        case 0:  src = qf; gamma = qg; beta = qb; do_ln = true;  break;
        case 1:  src = kf; gamma = kg; beta = kb; do_ln = true;  break;
        default: src = vf; gamma = qg; beta = qb; do_ln = false; break;
    }

    const float* vsrc = src + (size_t)bn * DM + h * DH;
    float x0 = vsrc[lane], x1 = vsrc[lane + 32];
    float o0 = x0, o1 = x1;
    if (do_ln) {
        float s = x0 + x1, s2 = x0 * x0 + x1 * x1;
#pragma unroll
        for (int m = 16; m >= 1; m >>= 1) {
            s  += __shfl_xor(s,  m, 32);
            s2 += __shfl_xor(s2, m, 32);
        }
        float mu  = s * (1.0f / DH);
        float var = s2 * (1.0f / DH) - mu * mu;
        float rs  = rsqrtf(var + 1e-5f);
        o0 = (x0 - mu) * rs * gamma[lane]      + beta[lane];
        o1 = (x1 - mu) * rs * gamma[lane + 32] + beta[lane + 32];
    }
    if (blockIdx.y == 2) {  // V transposed: vT[bh][d][n]
        bf16_t* d = vtbf + (size_t)bh * DH * NSEQ + n;
        d[(size_t)lane * NSEQ]        = (bf16_t)o0;
        d[(size_t)(lane + 32) * NSEQ] = (bf16_t)o1;
    } else {
        bf16_t* d = (blockIdx.y == 0 ? qbf : kbf) + ((size_t)bh * NSEQ + n) * DH;
        d[lane]      = (bf16_t)o0;
        d[lane + 32] = (bf16_t)o1;
    }
}

// ---------------------------------------------------------------------------
// 3) Pair bias: bias[b,h,n,m] = pair[b,n,m,:] . Wpb[h,:]
//    Streams pair_repr (256 MB) exactly once via ASYNC global->LDS copies,
//    double-buffered against the WMMA consumer. Each 16-row m-tile is a
//    fully contiguous 8 KB block of pair_repr.
// ---------------------------------------------------------------------------
#define PB_TPW 4                 // tiles per wave
#define PB_WAVES 2               // waves per block (2 x 2 x 8KB = 32KB LDS)

__device__ __forceinline__ void pb_stage_async(const float* gsrc, float* ldst, int lane)
{
    unsigned lbase = (unsigned)(uintptr_t)ldst + (unsigned)lane * 16u;
    unsigned long long gbase = (unsigned long long)(uintptr_t)gsrc
                             + (unsigned long long)lane * 16ull;
#pragma unroll
    for (int i = 0; i < 16; ++i) {       // 16 x (32 lanes x 16B) = 8KB
        unsigned           l = lbase + (unsigned)i * 512u;
        unsigned long long a = gbase + (unsigned long long)i * 512ull;
        asm volatile("global_load_async_to_lds_b128 %0, %1, off"
                     :: "v"(l), "v"(a) : "memory");
    }
}

__global__ void pba_pair_bias(const float* __restrict__ pair,
                              const float* __restrict__ Wpb,
                              float* __restrict__ bias)
{
    __shared__ float stage[PB_WAVES][2][16 * DP];   // 32 KB

    const int wid  = threadIdx.x >> 5;
    const int lane = threadIdx.x & 31;
    const int l16  = lane & 15;
    const int g    = lane >> 4;

    const int MT = NSEQ / 16;                        // 32 m-tiles per (b,n)
    int wave = blockIdx.x * PB_WAVES + wid;          // 8192 waves
    int t0   = wave * PB_TPW;                        // tiles 0..32767
    if (t0 >= BATCH * NSEQ * MT) return;

    // Wpb B-fragments are tile-invariant: hoist into registers (4 p-chunks)
    v16bf wb[4];
#pragma unroll
    for (int c = 0; c < 4; ++c)
#pragma unroll
        for (int e = 0; e < 16; ++e)
            wb[c][e] = (l16 < NH) ? (bf16_t)Wpb[l16 * DP + c * 32 + kmap(e, g)]
                                  : (bf16_t)0.0f;

    float* buf[2] = { stage[wid][0], stage[wid][1] };

    // prime the pipeline: tile t0 -> buf0 (tile t is contiguous at pair + t*2048)
    pb_stage_async(pair + (size_t)t0 * (16 * DP), buf[0], lane);

#pragma unroll
    for (int i = 0; i < PB_TPW; ++i) {
        int t = t0 + i;
        float* cur = buf[i & 1];
        if (i + 1 < PB_TPW) {
            pb_stage_async(pair + (size_t)(t + 1) * (16 * DP), buf[(i + 1) & 1], lane);
            asm volatile("s_wait_asynccnt 0x10" ::: "memory");  // cur's 16 retired
        } else {
            asm volatile("s_wait_asynccnt 0x0" ::: "memory");
        }

        const float* prow = cur + l16 * DP;          // this lane's m-row
        v8f acc = {};
#pragma unroll
        for (int c = 0; c < 4; ++c) {
            v16bf a;
#pragma unroll
            for (int e = 0; e < 16; ++e)
                a[e] = (bf16_t)prow[c * 32 + kmap(e, g)];
            acc = __builtin_amdgcn_wmma_f32_16x16x32_bf16(
                      false, a, false, wb[c], (short)0, acc, false, false);
        }

        int m0 = (t % MT) * 16;
        int bn = t / MT;
        int n  = bn % NSEQ, b = bn / NSEQ;
        if (l16 < NH) {  // C tile: row = local m = r+8g, col = head = l16
            float* brow = bias + (((size_t)(b * NH + l16) * NSEQ + n) * NSEQ)
                        + m0 + 8 * g;
#pragma unroll
            for (int r = 0; r < 8; ++r) brow[r] = acc[r];
        }
    }
}

// ---------------------------------------------------------------------------
// 4) Flash attention partials: each wave = (16-row q tile) x (128-wide m split).
//    QK^T + bias + mask, online softmax in f32, P@V — bf16 WMMA throughout.
//    Writes unnormalized acc + running (max, sum) per row.
// ---------------------------------------------------------------------------
#define MSPLIT 4
#define MSPAN  (NSEQ / MSPLIT)   // 128

__global__ void pba_attention(const bf16_t* __restrict__ qbf,
                              const bf16_t* __restrict__ kbf,
                              const bf16_t* __restrict__ vtbf,
                              const float* __restrict__ bias,
                              const unsigned char* __restrict__ mask,
                              float* __restrict__ pacc,
                              float* __restrict__ pml)
{
    __shared__ bf16_t pshare[4][16 * 32];

    const int wid  = threadIdx.x >> 5;
    const int lane = threadIdx.x & 31;
    const int l16  = lane & 15;
    const int g    = lane >> 4;

    int t     = blockIdx.x * 4 + wid;        // 2048 = 512 tiles x 4 splits
    int split = t & (MSPLIT - 1);
    int tile  = t >> 2;                      // tile = bh*32 + nt
    int nt    = tile & 31;
    int bh    = tile >> 5;
    int b     = bh >> 3;
    const int n0 = nt * 16;

    const bf16_t* qh = qbf  + (size_t)bh * NSEQ * DH;
    const bf16_t* kh = kbf  + (size_t)bh * NSEQ * DH;
    const bf16_t* vt = vtbf + (size_t)bh * DH * NSEQ;
    const float*  bb = bias + (size_t)bh * NSEQ * NSEQ + (size_t)n0 * NSEQ;
    const unsigned char* mb = mask + (size_t)b * NSEQ;
    const float scale = 0.125f;              // 1/sqrt(DH)
    const float NEG_INF = -__builtin_inff();

    v16bf qa[2];
#pragma unroll
    for (int c = 0; c < 2; ++c)
#pragma unroll
        for (int e = 0; e < 16; ++e)
            qa[c][e] = qh[(size_t)(n0 + l16) * DH + c * 32 + kmap(e, g)];

    v8f zero = {};
    v8f oacc[4];
#pragma unroll
    for (int dc = 0; dc < 4; ++dc) oacc[dc] = zero;

    float mrun[8], lrun[8];
#pragma unroll
    for (int r = 0; r < 8; ++r) { mrun[r] = NEG_INF; lrun[r] = 0.0f; }

    bf16_t* pl = pshare[wid];

    const int mbeg = split * MSPAN;
    for (int m0 = mbeg; m0 < mbeg + MSPAN; m0 += 32) {
        float lg[2][8];
#pragma unroll
        for (int s = 0; s < 2; ++s) {
            v8f acc = zero;
#pragma unroll
            for (int c = 0; c < 2; ++c) {
                v16bf kfrag;
#pragma unroll
                for (int e = 0; e < 16; ++e)
                    kfrag[e] = kh[(size_t)(m0 + s * 16 + l16) * DH + c * 32 + kmap(e, g)];
                acc = __builtin_amdgcn_wmma_f32_16x16x32_bf16(
                          false, qa[c], false, kfrag, (short)0, acc, false, false);
            }
            int  m  = m0 + s * 16 + l16;
            bool mk = mb[m] != 0;
#pragma unroll
            for (int r = 0; r < 8; ++r) {
                float v = acc[r] * scale + bb[(size_t)(r + 8 * g) * NSEQ + m];
                lg[s][r] = mk ? v : NEG_INF;
            }
        }

        float alpha[8];
#pragma unroll
        for (int r = 0; r < 8; ++r) {
            float rmax = fmaxf(lg[0][r], lg[1][r]);
#pragma unroll
            for (int mm = 8; mm >= 1; mm >>= 1)
                rmax = fmaxf(rmax, __shfl_xor(rmax, mm, 32));
            float mnew = fmaxf(mrun[r], rmax);
            float a_   = (mnew == NEG_INF) ? 1.0f : __expf(mrun[r] - mnew);
            float p0   = (lg[0][r] == NEG_INF) ? 0.0f : __expf(lg[0][r] - mnew);
            float p1   = (lg[1][r] == NEG_INF) ? 0.0f : __expf(lg[1][r] - mnew);
            float rs   = p0 + p1;
#pragma unroll
            for (int mm = 8; mm >= 1; mm >>= 1)
                rs += __shfl_xor(rs, mm, 32);
            lrun[r]  = lrun[r] * a_ + rs;
            mrun[r]  = mnew;
            alpha[r] = a_;
            lg[0][r] = p0;
            lg[1][r] = p1;
        }
#pragma unroll
        for (int dc = 0; dc < 4; ++dc)
#pragma unroll
            for (int r = 0; r < 8; ++r)
                oacc[dc][r] *= alpha[r];

        // C-layout -> A-fragment relayout through LDS
        __syncthreads();
#pragma unroll
        for (int s = 0; s < 2; ++s)
#pragma unroll
            for (int r = 0; r < 8; ++r)
                pl[(r + 8 * g) * 32 + s * 16 + l16] = (bf16_t)lg[s][r];
        __syncthreads();

        v16bf pa;
#pragma unroll
        for (int e = 0; e < 16; ++e)
            pa[e] = pl[l16 * 32 + kmap(e, g)];

        // P @ V with transposed V: contiguous along m
#pragma unroll
        for (int dc = 0; dc < 4; ++dc) {
            v16bf vfrag;
#pragma unroll
            for (int e = 0; e < 16; ++e)
                vfrag[e] = vt[(size_t)(dc * 16 + l16) * NSEQ + m0 + kmap(e, g)];
            oacc[dc] = __builtin_amdgcn_wmma_f32_16x16x32_bf16(
                           false, pa, false, vfrag, (short)0, oacc[dc], false, false);
        }
    }

    // store unnormalized partials: pacc[t][row16][d64], pml[t][row16][{m,l}]
    float* pa_out = pacc + (size_t)t * (16 * DH);
#pragma unroll
    for (int dc = 0; dc < 4; ++dc)
#pragma unroll
        for (int r = 0; r < 8; ++r)
            pa_out[(r + 8 * g) * DH + dc * 16 + l16] = oacc[dc][r];
    if (l16 == 0) {
#pragma unroll
        for (int r = 0; r < 8; ++r) {
            pml[(size_t)t * 32 + (r + 8 * g) * 2 + 0] = mrun[r];
            pml[(size_t)t * 32 + (r + 8 * g) * 2 + 1] = lrun[r];
        }
    }
}

// ---------------------------------------------------------------------------
// 4b) Combine the 4 m-splits (log-sum-exp merge) -> attn_out bf16 [B,N,H*DH]
// ---------------------------------------------------------------------------
__global__ void pba_combine(const float* __restrict__ pacc,
                            const float* __restrict__ pml,
                            bf16_t* __restrict__ attn_out)
{
    int idx = blockIdx.x * blockDim.x + threadIdx.x;     // B*H*N*DH = 524288
    if (idx >= BATCH * NH * NSEQ * DH) return;
    int d    = idx & (DH - 1);
    int rg   = idx >> 6;          // global row 0..8191
    int tile = rg >> 4;
    int r    = rg & 15;

    const float NEG_INF = -__builtin_inff();
    float mloc[MSPLIT], lloc[MSPLIT];
    float mf = NEG_INF;
#pragma unroll
    for (int c = 0; c < MSPLIT; ++c) {
        mloc[c] = pml[(size_t)(tile * MSPLIT + c) * 32 + r * 2 + 0];
        lloc[c] = pml[(size_t)(tile * MSPLIT + c) * 32 + r * 2 + 1];
        mf = fmaxf(mf, mloc[c]);
    }
    float out = 0.0f, l = 0.0f;
    if (mf != NEG_INF) {
#pragma unroll
        for (int c = 0; c < MSPLIT; ++c) {
            float w = __expf(mloc[c] - mf);
            out += w * pacc[(size_t)(tile * MSPLIT + c) * (16 * DH) + r * DH + d];
            l   += w * lloc[c];
        }
    }
    float val = (l > 0.0f) ? out / l : 0.0f;             // NaN guard (masked rows)

    int nt = tile & 31, bh = tile >> 5;
    int h  = bh & 7,    b  = bh >> 3;
    int n  = nt * 16 + r;
    attn_out[((size_t)(b * NSEQ) + n) * DM + h * DH + d] = (bf16_t)val;
}

// ---------------------------------------------------------------------------
// 5) Output projection: out = attn(bf16) @ Wo^T -> f32
// ---------------------------------------------------------------------------
__global__ void pba_out_proj(const bf16_t* __restrict__ ao,
                             const float* __restrict__ Wo,
                             float* __restrict__ out)
{
    const int wid  = threadIdx.x >> 5;
    const int lane = threadIdx.x & 31;
    const int l16  = lane & 15;
    const int g    = lane >> 4;

    const int TILES_J = DM / 16;
    int t  = blockIdx.x * (blockDim.x >> 5) + wid;
    int ti = t / TILES_J, tj = t % TILES_J;
    if (ti >= (BATCH * NSEQ) / 16) return;

    v8f acc = {};
    for (int k0 = 0; k0 < DM; k0 += 32) {
        v16bf a, b;
#pragma unroll
        for (int e = 0; e < 16; ++e) {
            a[e] = ao[(size_t)(ti * 16 + l16) * DM + k0 + kmap(e, g)];
            b[e] = (bf16_t)Wo[(size_t)(tj * 16 + l16) * DM + k0 + kmap(e, g)];
        }
        acc = __builtin_amdgcn_wmma_f32_16x16x32_bf16(
                  false, a, false, b, (short)0, acc, false, false);
    }
#pragma unroll
    for (int r = 0; r < 8; ++r)
        out[(size_t)(ti * 16 + r + 8 * g) * DM + tj * 16 + l16] = acc[r];
}

// ---------------------------------------------------------------------------
extern "C" void kernel_launch(void* const* d_in, const int* in_sizes, int n_in,
                              void* d_out, int out_size, void* d_ws, size_t ws_size,
                              hipStream_t stream) {
    (void)in_sizes; (void)n_in; (void)out_size; (void)ws_size;

    const float*         x    = (const float*)d_in[0];
    const float*         pair = (const float*)d_in[1];
    const unsigned char* mask = (const unsigned char*)d_in[2];
    const float*         Wq   = (const float*)d_in[3];
    const float*         Wk   = (const float*)d_in[4];
    const float*         Wv   = (const float*)d_in[5];
    const float*         Wo   = (const float*)d_in[6];
    const float*         qg   = (const float*)d_in[7];
    const float*         qb   = (const float*)d_in[8];
    const float*         kg   = (const float*)d_in[9];
    const float*         kb   = (const float*)d_in[10];
    const float*         Wpb  = (const float*)d_in[11];

    const size_t nTok   = (size_t)BATCH * NSEQ;          // 1024
    const size_t szProj = nTok * DM;                     // 524288 elems
    const int    nTiles = BATCH * NH * (NSEQ / 16);      // 512
    char* w = (char*)d_ws;
    float*  qf   = (float*)w;  w += szProj * sizeof(float);
    float*  kf   = (float*)w;  w += szProj * sizeof(float);
    float*  vf   = (float*)w;  w += szProj * sizeof(float);
    bf16_t* qbf  = (bf16_t*)w; w += szProj * sizeof(bf16_t);
    bf16_t* kbf  = (bf16_t*)w; w += szProj * sizeof(bf16_t);
    bf16_t* vtbf = (bf16_t*)w; w += szProj * sizeof(bf16_t);
    bf16_t* aout = (bf16_t*)w; w += szProj * sizeof(bf16_t);
    w = (char*)(((uintptr_t)w + 255) & ~(uintptr_t)255);
    float*  bias = (float*)w;  w += (size_t)BATCH * NH * NSEQ * NSEQ * sizeof(float);
    float*  pacc = (float*)w;  w += (size_t)nTiles * MSPLIT * 16 * DH * sizeof(float);
    float*  pml  = (float*)w;  // nTiles * MSPLIT * 32 floats

    dim3 blk(128);

    // 1) QKV GEMMs: 2048 tiles each, z selects q/k/v
    pba_qkv_gemm<<<dim3(512, 1, 3), blk, 0, stream>>>(x, Wq, Wk, Wv, qf, kf, vf);

    // 2) LayerNorm + pack (y selects q/k/v); V written transposed
    pba_ln_pack<<<dim3(2048, 3, 1), blk, 0, stream>>>(qf, kf, vf, qg, qb, kg, kb,
                                                      qbf, kbf, vtbf);

    // 3) Pair bias: 8192 waves x 4 tiles, async double-buffered LDS staging
    pba_pair_bias<<<dim3(4096, 1, 1), dim3(32 * PB_WAVES), 0, stream>>>(pair, Wpb, bias);

    // 4) Flash attention partials: 512 tiles x 4 m-splits = 2048 waves
    pba_attention<<<dim3(512, 1, 1), blk, 0, stream>>>(qbf, kbf, vtbf, bias, mask,
                                                       pacc, pml);

    // 4b) Split combine
    pba_combine<<<dim3(2048, 1, 1), dim3(256), 0, stream>>>(pacc, pml, aout);

    // 5) Output projection
    pba_out_proj<<<dim3(512, 1, 1), blk, 0, stream>>>(aout, Wo, (float*)d_out);
}
